// GlobalAggregator_79860621902396
// MI455X (gfx1250) — matrix operand order repaired and checked by reading, not compiled, not run on hardware
//
#include <hip/hip_runtime.h>
#include <stdint.h>

// ---------------------------------------------------------------------------
// GlobalAggregator for MI455X (gfx1250, wave32, WMMA)
//   B=512, N=256, D=128
//   Phase 1: score[b,i] = relu([h*s, h] @ [w0;w1] + bias) . a0   (bf16 WMMA)
//   Phase 2: out[b,i,:] = score[b,i] * sum_j (adj==1) * h[b,j,:] (bf16 WMMA)
// B-operands are converted to bf16 ONCE per block into LDS (padded stride).
// Inner loop: per K-step, clause-load the whole 256B LDS row (8 B-fragments)
// then issue 8 WMMAs -> one dscnt wait per 8 WMMAs instead of per WMMA.
// ---------------------------------------------------------------------------

typedef __attribute__((ext_vector_type(16))) __bf16    v16bf;
typedef __attribute__((ext_vector_type(8)))  float     v8f;
typedef __attribute__((ext_vector_type(4)))  float     v4f;
typedef __attribute__((ext_vector_type(2)))  float     v2f;
typedef __attribute__((ext_vector_type(4)))  int       v4i;
typedef __attribute__((ext_vector_type(4)))  uint32_t  v4u;

union Frag { v16bf v; uint32_t u[8]; v4u q[2]; };

// LDS tile: 256 rows x 64 dwords (128 bf16), padded to 68 dwords/row so the
// row-strided B-fragment reads hit distinct banks (68*4B keeps 16B alignment).
#define LDS_STRIDE 68
#define LDS_DWORDS (256 * LDS_STRIDE)
#define LDS_BYTES  (LDS_DWORDS * 4)

// ---- f32 pair -> packed bf16 (RNE): one VALU instruction -------------------
__device__ __forceinline__ uint32_t pack_bf16(float a, float b) {
  uint32_t r;
  asm("v_cvt_pk_bf16_f32 %0, %1, %2" : "=v"(r) : "v"(a), "v"(b));
  return r;
}

// adjacency {0,1} -> packed bf16 pair: (a + (b<<16)) * 0x3F80 has no carries,
// yielding (a?0x3F80:0) | (b?0x3F800000:0) in 2 VALU ops (bf16 1.0 = 0x3F80).
__device__ __forceinline__ uint32_t pack_adj(int a, int b) {
  return (uint32_t)(a + (b << 16)) * 0x3F80u;
}

// ---------------------------------------------------------------------------
// Kernel 1: scores. 256 threads = 8 waves, one 16-row tile per wave.
// grid = BN/128 = 1024 blocks.  score -> d_ws (BN floats)
// ---------------------------------------------------------------------------
__global__ __launch_bounds__(256) void score_kernel(
    const float* __restrict__ h, const float* __restrict__ s,
    const float* __restrict__ w0, const float* __restrict__ w1,
    const float* __restrict__ a0, const float* __restrict__ bias,
    float* __restrict__ score)
{
  extern __shared__ uint32_t lds_w[];   // [256][LDS_STRIDE] bf16 pairs of [w0;w1]

  // ---- cooperative one-time convert of W = [w0;w1] (256x128 f32 -> bf16) ---
  for (int p = threadIdx.x; p < 256 * 64; p += 256) {
    const int k = p >> 6, c = p & 63;
    const float* wrow = (k < 128) ? (w0 + (size_t)k * 128)
                                  : (w1 + (size_t)(k - 128) * 128);
    const v2f f = ((const v2f*)wrow)[c];
    lds_w[k * LDS_STRIDE + c] = pack_bf16(f.x, f.y);
  }
  __syncthreads();

  const int lane = threadIdx.x & 31;
  const int wave = threadIdx.x >> 5;
  const int l16  = lane & 15;
  const int half = lane >> 4;
  const int rowbase = blockIdx.x * 128 + wave * 16;

  // ---- A fragments: this lane's row of X = [h*s (K 0..127), h (K 128..255)]
  // Per K-step the lane's packed pairs form two contiguous 32B runs:
  //   run0 = kb + 8*half + {0..7},  run1 = kb + 16 + 8*half + {0..7}
  const size_t rowoff = (size_t)(rowbase + l16) * 128;
  Frag afr[8];
#pragma unroll
  for (int t = 0; t < 8; ++t) {
    const int kb = t * 32;
    const int o0 = (kb & 127) + 8 * half;
    v4f a0v = *(const v4f*)(h + rowoff + o0);
    v4f a1v = *(const v4f*)(h + rowoff + o0 + 4);
    v4f b0v = *(const v4f*)(h + rowoff + o0 + 16);
    v4f b1v = *(const v4f*)(h + rowoff + o0 + 20);
    if (kb < 128) {     // multiply by session_info for the first K half
      a0v *= *(const v4f*)(s + rowoff + o0);
      a1v *= *(const v4f*)(s + rowoff + o0 + 4);
      b0v *= *(const v4f*)(s + rowoff + o0 + 16);
      b1v *= *(const v4f*)(s + rowoff + o0 + 20);
    }
    afr[t].u[0] = pack_bf16(a0v.x, a0v.y);
    afr[t].u[1] = pack_bf16(a0v.z, a0v.w);
    afr[t].u[2] = pack_bf16(a1v.x, a1v.y);
    afr[t].u[3] = pack_bf16(a1v.z, a1v.w);
    afr[t].u[4] = pack_bf16(b0v.x, b0v.y);
    afr[t].u[5] = pack_bf16(b0v.z, b0v.w);
    afr[t].u[6] = pack_bf16(b1v.x, b1v.y);
    afr[t].u[7] = pack_bf16(b1v.z, b1v.w);
  }

  // ---- accumulate: K-step outer, N-tile inner; full LDS row per K-step ----
  v8f acc[8] = {{}, {}, {}, {}, {}, {}, {}, {}};
#pragma unroll
  for (int t = 0; t < 8; ++t) {
    const unsigned short* p = (const unsigned short*)(lds_w + (t * 32 + 8 * half) * LDS_STRIDE);   // B column n = nt*16+l16, K in table order
    Frag brow[8];
#pragma unroll
    for (int nt = 0; nt < 8; ++nt) {
#pragma unroll
      for (int e = 0; e < 8; ++e) {
        const int kk = (e < 4) ? 2 * e : 2 * e + 8;
        brow[nt].u[e] = (uint32_t)p[kk * 2 * LDS_STRIDE + nt * 16 + l16] | ((uint32_t)p[(kk + 1) * 2 * LDS_STRIDE + nt * 16 + l16] << 16);
      }
    }
#pragma unroll
    for (int nt = 0; nt < 8; ++nt)
      acc[nt] = __builtin_amdgcn_wmma_f32_16x16x32_bf16(
          false, afr[t].v, false, brow[nt].v, (short)0, acc[nt], false, false);
  }

  // ---- epilogue: relu(acc + bias[d]) * a0[d], then row-dot reduction ------
  float partial[8] = {0.f, 0.f, 0.f, 0.f, 0.f, 0.f, 0.f, 0.f};
#pragma unroll
  for (int nt = 0; nt < 8; ++nt) {
    const int   d  = nt * 16 + l16;
    const float bd = bias[d];
    const float ad = a0[d];
#pragma unroll
    for (int r = 0; r < 8; ++r) {
      float val = acc[nt][r] + bd;
      val = val > 0.f ? val : 0.f;
      partial[r] = fmaf(val, ad, partial[r]);
    }
  }

  // width-16 butterfly: rows 0-7 reduce over lanes 0-15, rows 8-15 over 16-31
#pragma unroll
  for (int off = 8; off >= 1; off >>= 1) {
#pragma unroll
    for (int r = 0; r < 8; ++r)
      partial[r] += __shfl_xor(partial[r], off, 16);
  }
  if (l16 == 0) {
#pragma unroll
    for (int r = 0; r < 8; ++r)
      score[rowbase + 8 * half + r] = partial[r];
  }
}

// ---------------------------------------------------------------------------
// Kernel 2: out[row,:] = score[row] * (adj_tile @ h_b).
// One block per batch: 512 threads = 16 waves, one 16-row i-tile per wave.
// h_b is converted to bf16 in LDS exactly once per batch.
// ---------------------------------------------------------------------------
__global__ __launch_bounds__(512) void agg_kernel(
    const float* __restrict__ h, const int* __restrict__ adj,
    const float* __restrict__ score, float* __restrict__ out)
{
  extern __shared__ uint32_t lds_h[];   // [256][LDS_STRIDE] bf16 pairs of h_b

  const int b = blockIdx.x;
  const float* hb = h + (size_t)b * 256 * 128;

  // ---- cooperative one-time convert of h_b (256x128 f32 -> bf16) ----------
  for (int p = threadIdx.x; p < 256 * 64; p += 512) {
    const int j = p >> 6, c = p & 63;
    const v2f f = ((const v2f*)(hb + (size_t)j * 128))[c];
    lds_h[j * LDS_STRIDE + c] = pack_bf16(f.x, f.y);
  }

  const int lane = threadIdx.x & 31;
  const int wave = threadIdx.x >> 5;
  const int l16  = lane & 15;
  const int half = lane >> 4;
  const int i0   = wave * 16;

  const int* arow = adj + ((size_t)b * 256 + (i0 + l16)) * 256;
  __builtin_prefetch(arow, 0, 1);          // stream in this lane's adjacency row
  __builtin_prefetch(arow + 128, 0, 1);

  __syncthreads();

  // ---- A fragments: adjacency 0/1 -> bf16 (exact), two 32B runs per K-step
  Frag afr[8];
#pragma unroll
  for (int t = 0; t < 8; ++t) {
    const int kb = t * 32;
    const v4i r0 = *(const v4i*)(arow + kb + 8 * half);
    const v4i r1 = *(const v4i*)(arow + kb + 8 * half + 4);
    const v4i r2 = *(const v4i*)(arow + kb + 16 + 8 * half);
    const v4i r3 = *(const v4i*)(arow + kb + 16 + 8 * half + 4);
    afr[t].u[0] = pack_adj(r0.x, r0.y);
    afr[t].u[1] = pack_adj(r0.z, r0.w);
    afr[t].u[2] = pack_adj(r1.x, r1.y);
    afr[t].u[3] = pack_adj(r1.z, r1.w);
    afr[t].u[4] = pack_adj(r2.x, r2.y);
    afr[t].u[5] = pack_adj(r2.z, r2.w);
    afr[t].u[6] = pack_adj(r3.x, r3.y);
    afr[t].u[7] = pack_adj(r3.z, r3.w);
  }

  // ---- accumulate: K-step outer, N-tile inner; full LDS row per K-step ----
  v8f acc[8] = {{}, {}, {}, {}, {}, {}, {}, {}};
#pragma unroll
  for (int t = 0; t < 8; ++t) {
    const unsigned short* p = (const unsigned short*)(lds_h + (t * 32 + 8 * half) * LDS_STRIDE);   // B column n = nt*16+l16, K in table order
    Frag brow[8];
#pragma unroll
    for (int nt = 0; nt < 8; ++nt) {
#pragma unroll
      for (int e = 0; e < 8; ++e) {
        const int kk = (e < 4) ? 2 * e : 2 * e + 8;
        brow[nt].u[e] = (uint32_t)p[kk * 2 * LDS_STRIDE + nt * 16 + l16] | ((uint32_t)p[(kk + 1) * 2 * LDS_STRIDE + nt * 16 + l16] << 16);
      }
    }
#pragma unroll
    for (int nt = 0; nt < 8; ++nt)
      acc[nt] = __builtin_amdgcn_wmma_f32_16x16x32_bf16(
          false, afr[t].v, false, brow[nt].v, (short)0, acc[nt], false, false);
  }

  // ---- epilogue: scale by score[row] and store -----------------------------
  float sc[8];
#pragma unroll
  for (int r = 0; r < 8; ++r)
    sc[r] = score[(size_t)b * 256 + i0 + 8 * half + r];

#pragma unroll
  for (int nt = 0; nt < 8; ++nt) {
#pragma unroll
    for (int r = 0; r < 8; ++r) {
      const size_t row = (size_t)b * 256 + i0 + 8 * half + r;
      out[row * 128 + nt * 16 + l16] = acc[nt][r] * sc[r];
    }
  }
}

// ---------------------------------------------------------------------------
extern "C" void kernel_launch(void* const* d_in, const int* in_sizes, int n_in,
                              void* d_out, int out_size, void* d_ws, size_t ws_size,
                              hipStream_t stream) {
  (void)in_sizes; (void)n_in; (void)out_size; (void)ws_size;
  const float* h    = (const float*)d_in[0];
  const float* s    = (const float*)d_in[1];
  const float* w0   = (const float*)d_in[2];
  const float* w1   = (const float*)d_in[3];
  const float* a0   = (const float*)d_in[4];
  const float* bias = (const float*)d_in[5];
  const int*   adj  = (const int*)d_in[6];
  float* score = (float*)d_ws;          // BN = 131072 floats (512 KB)
  float* out   = (float*)d_out;

  // Phase 1: BN = 131072 rows, 128 rows per 8-wave block -> 1024 blocks
  score_kernel<<<1024, 256, LDS_BYTES, stream>>>(h, s, w0, w1, a0, bias, score);
  // Phase 2: one block per batch (16 waves cover the 256 rows)
  agg_kernel<<<512, 512, LDS_BYTES, stream>>>(h, adj, score, out);
}
